// DotProductAttentionTransformerMD17_Serial_44212393345456
// MI455X (gfx1250) — matrix operand order repaired and checked by reading, not compiled, not run on hardware
//
#include <hip/hip_runtime.h>
#include <hip/hip_bf16.h>

// ---------------------------------------------------------------------------
// CDNA5 (gfx1250) implementation, round 4:
//  - node_vec and m_s_node staged to bf16 once -> all big WMMA kernels now use
//    conversion-free b128 A-fragment loads (no cvt chains / hazard v_nops)
//  - t0/t4/t5 fused into one tri-output GEMM (A read once, 3 WMMAs per A frag)
// ---------------------------------------------------------------------------

typedef __attribute__((ext_vector_type(16))) __bf16 v16bf;
typedef __attribute__((ext_vector_type(8)))  __bf16 v8bf;
typedef __attribute__((ext_vector_type(4)))  __bf16 v4bf;
typedef __attribute__((ext_vector_type(8)))  float  v8f;

#define H_DIM 128

union frag16 {
  v16bf v;
  struct { v8bf lo, hi; } h;
};

// ---- A fragment (16x32) from fp32 activations -----------------------------
__device__ __forceinline__ v16bf load_a_f32(const float* __restrict__ rowp,
                                            int kk, int half) {
  v16bf a;
  const float* p = rowp + kk + half * 8;
#pragma unroll
  for (int e = 0; e < 8; ++e) a[e] = (__bf16)p[e];
#pragma unroll
  for (int e = 0; e < 8; ++e) a[8 + e] = (__bf16)p[16 + e];
  return a;
}

// ---- A fragment from bf16 activations: two b128 loads ---------------------
__device__ __forceinline__ v16bf load_a_bf(const __bf16* __restrict__ rowp,
                                           int kk, int half) {
  const __bf16* p = rowp + kk + half * 8;
  frag16 f;
  f.h.lo = *(const v8bf*)(p);
  f.h.hi = *(const v8bf*)(p + 16);
  return f.v;
}

// ---- B fragment from pre-packed weights: two b128 loads -------------------
__device__ __forceinline__ v16bf load_b_packed(const __bf16* __restrict__ P,
                                               int tilesN, int kk, int ntile,
                                               int lane) {
  const __bf16* p = P + (((size_t)(kk >> 5) * tilesN + ntile) * 32 + lane) * 16;
  frag16 f;
  f.h.lo = *(const v8bf*)(p);
  f.h.hi = *(const v8bf*)(p + 8);
  return f.v;
}

__device__ __forceinline__ float silu_f(float x) {
  return x * (1.0f / (1.0f + __expf(-x)));
}

__device__ __forceinline__ float4 ld4bf(const __bf16* __restrict__ p) {
  v4bf t = *(const v4bf*)(p);
  return make_float4((float)t[0], (float)t[1], (float)t[2], (float)t[3]);
}

__device__ __forceinline__ void st4bf(__bf16* __restrict__ p, float4 v) {
  v4bf t;
  t[0] = (__bf16)v.x; t[1] = (__bf16)v.y; t[2] = (__bf16)v.z; t[3] = (__bf16)v.w;
  *(v4bf*)p = t;
}

// ---- bulk f32 -> bf16 staging (8 elements / thread) -----------------------
__global__ void cvt_f32_bf16_kernel(const float* __restrict__ src,
                                    __bf16* __restrict__ dst, size_t n8) {
  const size_t i = (size_t)blockIdx.x * blockDim.x + threadIdx.x;
  if (i >= n8) return;
  const float4* s = (const float4*)src + i * 2;
  const float4 x = s[0], y = s[1];
  v8bf o;
  o[0] = (__bf16)x.x; o[1] = (__bf16)x.y; o[2] = (__bf16)x.z; o[3] = (__bf16)x.w;
  o[4] = (__bf16)y.x; o[5] = (__bf16)y.y; o[6] = (__bf16)y.z; o[7] = (__bf16)y.w;
  ((v8bf*)dst)[i] = o;
}

// ---- one-time weight pack: W[K,Nout] f32 -> WMMA B-frag layout bf16 -------
__global__ void pack_w_kernel(const float* __restrict__ W, __bf16* __restrict__ P,
                              int K, int Nout) {
  const int wid  = (int)((blockIdx.x * blockDim.x + threadIdx.x) >> 5);
  const int lane = (int)(threadIdx.x & 31);
  const int tilesN = Nout >> 4;
  const int total = (K >> 5) * tilesN;
  if (wid >= total) return;
  const int ks = wid / tilesN, tn = wid - ks * tilesN;
  const int col = tn * 16 + (lane & 15), half = lane >> 4;
  const float* src = W + (size_t)(ks * 32 + half * 8) * Nout + col;
  __bf16* dst = P + ((size_t)wid * 32 + lane) * 16;
#pragma unroll
  for (int e = 0; e < 16; ++e) dst[e] = (__bf16)src[(size_t)(e + (e & 8)) * Nout];
}

// ---- GEMM: C = op(A @ W + bias). ROW2: 32x64 strip/wave, else 16x64 -------
template <bool ROW2, bool A_BF16, bool C_BF16, bool BIAS, bool SILU>
__global__ void wmma_gemm4_kernel(const void* __restrict__ Av,
                                  const __bf16* __restrict__ Pw,
                                  const float* __restrict__ bias,
                                  void* __restrict__ Cv,
                                  int M, int K, int Nout) {
  const int wid  = (int)((blockIdx.x * blockDim.x + threadIdx.x) >> 5);
  const int lane = (int)(threadIdx.x & 31);
  const int tiles4 = Nout >> 6;                    // Nout is 64 or 128
  const int rshift = ROW2 ? 5 : 4;
  const int total = (M >> rshift) * tiles4;
  if (wid >= total) return;                        // wave-uniform guard
  const int tm = wid / tiles4, tg = wid - tm * tiles4;
  const int row0 = tm << rshift, colg = tg << 6;
  const int half = lane >> 4, lr = lane & 15;
  const int tilesN = Nout >> 4, nt0 = colg >> 4;

  v8f acc0[4] = {{}, {}, {}, {}};
  v8f acc1[4] = {{}, {}, {}, {}};
  const __bf16* arb0 = (const __bf16*)Av + (size_t)(row0 + lr) * K;
  const float*  arf0 = (const float*)Av + (size_t)(row0 + lr) * K;
  const __bf16* arb1 = arb0 + (size_t)16 * K;
  const float*  arf1 = arf0 + (size_t)16 * K;

  for (int kk = 0; kk < K; kk += 32) {
    v16bf a0, a1;
    if constexpr (A_BF16) {
      __builtin_prefetch(arb0 + kk + 32, 0, 1);
      a0 = load_a_bf(arb0, kk, half);
      if constexpr (ROW2) a1 = load_a_bf(arb1, kk, half);
    } else {
      __builtin_prefetch(arf0 + kk + 32, 0, 1);
      a0 = load_a_f32(arf0, kk, half);
      if constexpr (ROW2) a1 = load_a_f32(arf1, kk, half);
    }
#pragma unroll
    for (int j = 0; j < 4; ++j) {
      v16bf b = load_b_packed(Pw, tilesN, kk, nt0 + j, lane);
      acc0[j] = __builtin_amdgcn_wmma_f32_16x16x32_bf16(
          false, a0, false, b, (short)0, acc0[j], false, false);
      if constexpr (ROW2)
        acc1[j] = __builtin_amdgcn_wmma_f32_16x16x32_bf16(
            false, a1, false, b, (short)0, acc1[j], false, false);
    }
  }

#pragma unroll
  for (int j = 0; j < 4; ++j) {
    const int col = colg + j * 16 + lr;
    float bv = 0.0f;
    if (BIAS) bv = bias[col];
#pragma unroll
    for (int r = 0; r < 8; ++r) {
      float v = acc0[j][r] + bv;
      if (SILU) v = silu_f(v);
      const size_t idx = (size_t)(row0 + r + half * 8) * Nout + col;
      if constexpr (C_BF16) ((__bf16*)Cv)[idx] = (__bf16)v;
      else                  ((float*)Cv)[idx]  = v;
    }
    if constexpr (ROW2) {
#pragma unroll
      for (int r = 0; r < 8; ++r) {
        float v = acc1[j][r] + bv;
        if (SILU) v = silu_f(v);
        const size_t idx = (size_t)(row0 + 16 + r + half * 8) * Nout + col;
        if constexpr (C_BF16) ((__bf16*)Cv)[idx] = (__bf16)v;
        else                  ((float*)Cv)[idx]  = v;
      }
    }
  }
}

// ---- Fused tri-GEMM: T0/T4/T5 = A @ {L0,L4,L5}, A bf16 [M,128] ------------
// j-outer keeps live set ~40 VGPRs (out 24 + A 8 + B 8); A stays L1-hot.
__global__ void wmma_gemm_tri_kernel(const __bf16* __restrict__ A,
                                     const __bf16* __restrict__ P0,
                                     const __bf16* __restrict__ P4,
                                     const __bf16* __restrict__ P5,
                                     float* __restrict__ T0,
                                     float* __restrict__ T4,
                                     float* __restrict__ T5, int M) {
  const int wid  = (int)((blockIdx.x * blockDim.x + threadIdx.x) >> 5);
  const int lane = (int)(threadIdx.x & 31);
  const int tiles4 = H_DIM >> 6;                   // 2
  const int total = (M >> 4) * tiles4;
  if (wid >= total) return;
  const int tm = wid / tiles4, tg = wid - tm * tiles4;
  const int row0 = tm << 4, colg = tg << 6;
  const int half = lane >> 4, lr = lane & 15;
  const int tilesN = H_DIM >> 4, nt0 = colg >> 4;
  const __bf16* arow = A + (size_t)(row0 + lr) * H_DIM;

#pragma unroll 1
  for (int j = 0; j < 4; ++j) {
    v8f a0 = {};
    v8f a4 = {};
    v8f a5 = {};
#pragma unroll
    for (int kk = 0; kk < H_DIM; kk += 32) {
      v16bf af = load_a_bf(arow, kk, half);
      v16bf b0 = load_b_packed(P0, tilesN, kk, nt0 + j, lane);
      a0 = __builtin_amdgcn_wmma_f32_16x16x32_bf16(
          false, af, false, b0, (short)0, a0, false, false);
      v16bf b4 = load_b_packed(P4, tilesN, kk, nt0 + j, lane);
      a4 = __builtin_amdgcn_wmma_f32_16x16x32_bf16(
          false, af, false, b4, (short)0, a4, false, false);
      v16bf b5 = load_b_packed(P5, tilesN, kk, nt0 + j, lane);
      a5 = __builtin_amdgcn_wmma_f32_16x16x32_bf16(
          false, af, false, b5, (short)0, a5, false, false);
    }
    const int col = colg + j * 16 + lr;
#pragma unroll
    for (int r = 0; r < 8; ++r) {
      const size_t idx = (size_t)(row0 + r + half * 8) * H_DIM + col;
      T0[idx] = a0[r];
      T4[idx] = a4[r];
      T5[idx] = a5[r];
    }
  }
}

// ---- Fused: S[N,H] = sum_a (nv[:,a,:]@L2) * (nv[:,a,:]@L3), nv bf16 -------
__global__ void v1v2_wmma_kernel(const __bf16* __restrict__ nv,
                                 const __bf16* __restrict__ P2,
                                 const __bf16* __restrict__ P3,
                                 float* __restrict__ S, int Nn) {
  const int wid  = (int)((blockIdx.x * blockDim.x + threadIdx.x) >> 5);
  const int lane = (int)(threadIdx.x & 31);
  const int tiles4 = H_DIM >> 6;                   // 2
  const int total = (Nn >> 4) * tiles4;
  if (wid >= total) return;
  const int tm = wid / tiles4, tg = wid - tm * tiles4;
  const int row0 = tm << 4, colg = tg << 6;
  const int half = lane >> 4, lr = lane & 15;
  const int tilesN = H_DIM >> 4, nt0 = colg >> 4;

#pragma unroll 1
  for (int j = 0; j < 4; ++j) {
    v8f out = {};
#pragma unroll 1
    for (int a = 0; a < 3; ++a) {
      v8f acc1 = {};
      v8f acc2 = {};
      const __bf16* arow = nv + ((size_t)(row0 + lr) * 3 + a) * H_DIM;
#pragma unroll
      for (int kk = 0; kk < H_DIM; kk += 32) {
        v16bf af = load_a_bf(arow, kk, half);
        v16bf b2 = load_b_packed(P2, tilesN, kk, nt0 + j, lane);
        acc1 = __builtin_amdgcn_wmma_f32_16x16x32_bf16(
            false, af, false, b2, (short)0, acc1, false, false);
        v16bf b3 = load_b_packed(P3, tilesN, kk, nt0 + j, lane);
        acc2 = __builtin_amdgcn_wmma_f32_16x16x32_bf16(
            false, af, false, b3, (short)0, acc2, false, false);
      }
      out += acc1 * acc2;
    }
    const int col = colg + j * 16 + lr;
#pragma unroll
    for (int r = 0; r < 8; ++r)
      S[(size_t)(row0 + r + half * 8) * H_DIM + col] = out[r];
  }
}

// ---- GEMM w/ row-scale accumulate: Out[m,:] += R[m/3,:] * (A[m,:]@W) ------
__global__ void wmma_gemm_rowscale_add_kernel(const __bf16* __restrict__ A,
                                              const __bf16* __restrict__ Pw,
                                              const float* __restrict__ R,
                                              float* __restrict__ Out,
                                              int M, int K, int Nout) {
  const int wid  = (int)((blockIdx.x * blockDim.x + threadIdx.x) >> 5);
  const int lane = (int)(threadIdx.x & 31);
  const int tiles4 = Nout >> 6;
  const int total = (M >> 4) * tiles4;
  if (wid >= total) return;
  const int tm = wid / tiles4, tg = wid - tm * tiles4;
  const int row0 = tm << 4, colg = tg << 6;
  const int half = lane >> 4, lr = lane & 15;
  const int tilesN = Nout >> 4, nt0 = colg >> 4;

  v8f acc[4] = {{}, {}, {}, {}};
  const __bf16* arow = A + (size_t)(row0 + lr) * K;
  for (int kk = 0; kk < K; kk += 32) {
    v16bf a = load_a_bf(arow, kk, half);
#pragma unroll
    for (int j = 0; j < 4; ++j) {
      v16bf b = load_b_packed(Pw, tilesN, kk, nt0 + j, lane);
      acc[j] = __builtin_amdgcn_wmma_f32_16x16x32_bf16(
          false, a, false, b, (short)0, acc[j], false, false);
    }
  }
#pragma unroll
  for (int j = 0; j < 4; ++j) {
    const int col = colg + j * 16 + lr;
#pragma unroll
    for (int r = 0; r < 8; ++r) {
      const int row = row0 + r + half * 8;
      const float sc = R[(size_t)(row / 3) * Nout + col];
      const size_t idx = (size_t)row * Nout + col;
      Out[idx] += acc[j][r] * sc;    // dv finalized earlier (stream order)
    }
  }
}

// ---- Edge attention: attn=silu(dot/4); m_s=val*attn (bf16); scatter f32 ---
__global__ void edge_attn_kernel(const int* __restrict__ node_idx,
                                 const int* __restrict__ group_idx,
                                 const __bf16* __restrict__ q,
                                 const __bf16* __restrict__ k,
                                 const __bf16* __restrict__ val,
                                 const float* __restrict__ edge_attr,
                                 __bf16* __restrict__ m_s,
                                 float* __restrict__ m_s_node, int E) {
  const int e = (int)((blockIdx.x * blockDim.x + threadIdx.x) >> 5);
  if (e >= E) return;
  const int lane = (int)(threadIdx.x & 31);
  const int n = node_idx[e], g = group_idx[e];
  const int c0 = lane << 2;

  const float4 qv = ld4bf(q + (size_t)n * H_DIM + c0);
  const float4 kv = ld4bf(k + (size_t)g * H_DIM + c0);
  const float4 ev = *(const float4*)(edge_attr + (size_t)e * H_DIM + c0);
  float part = qv.x * kv.x * ev.x + qv.y * kv.y * ev.y +
               qv.z * kv.z * ev.z + qv.w * kv.w * ev.w;
  part += __shfl_xor(part, 1, 32);     // reduce 16-ch head over 4 lanes
  part += __shfl_xor(part, 2, 32);
  const float attn = silu_f(part * 0.25f);   // 1/sqrt(16)

  const float4 vv = ld4bf(val + (size_t)g * H_DIM + c0);
  const float4 ms = make_float4(vv.x * attn, vv.y * attn, vv.z * attn, vv.w * attn);
  st4bf(m_s + (size_t)e * H_DIM + c0, ms);

  float* dst = m_s_node + (size_t)n * H_DIM + c0;
  atomicAdd(dst + 0, ms.x);
  atomicAdd(dst + 1, ms.y);
  atomicAdd(dst + 2, ms.z);
  atomicAdd(dst + 3, ms.w);
}

// ---- Edge m_v scatter: dv[n,a,:] += sp*(-edge_vec[a]) + sv*group_vec ------
__global__ void edge_mv_kernel(const int* __restrict__ node_idx,
                               const int* __restrict__ group_idx,
                               const __bf16* __restrict__ sp,
                               const __bf16* __restrict__ sv,
                               const float* __restrict__ edge_vec,
                               const float* __restrict__ group_vec,
                               float* __restrict__ dv, int E) {
  const int e = (int)((blockIdx.x * blockDim.x + threadIdx.x) >> 5);
  if (e >= E) return;
  const int lane = (int)(threadIdx.x & 31);
  const int n = node_idx[e], g = group_idx[e];
  const int c0 = lane << 2;

  const float4 spv = ld4bf(sp + (size_t)e * H_DIM + c0);
  const float4 svv = ld4bf(sv + (size_t)e * H_DIM + c0);
#pragma unroll
  for (int a = 0; a < 3; ++a) {
    const float u = -edge_vec[(size_t)e * 3 + a];
    const float4 gv = *(const float4*)(group_vec + ((size_t)g * 3 + a) * H_DIM + c0);
    float* dst = dv + ((size_t)n * 3 + a) * H_DIM + c0;
    atomicAdd(dst + 0, spv.x * u + svv.x * gv.x);
    atomicAdd(dst + 1, spv.y * u + svv.y * gv.y);
    atomicAdd(dst + 2, spv.z * u + svv.z * gv.z);
    atomicAdd(dst + 3, spv.w * u + svv.w * gv.w);
  }
}

// ---- dx = s * t4 + t5 -----------------------------------------------------
__global__ void final_dx_kernel(const float* __restrict__ s,
                                const float* __restrict__ t4,
                                const float* __restrict__ t5,
                                float* __restrict__ dx, size_t n) {
  const size_t i = (size_t)blockIdx.x * blockDim.x + threadIdx.x;
  if (i < n) dx[i] = s[i] * t4[i] + t5[i];
}

__global__ void zero_kernel(float* __restrict__ p, size_t n) {
  const size_t i = (size_t)blockIdx.x * blockDim.x + threadIdx.x;
  if (i < n) p[i] = 0.0f;
}

// ---------------------------------------------------------------------------
extern "C" void kernel_launch(void* const* d_in, const int* in_sizes, int n_in,
                              void* d_out, int out_size, void* d_ws, size_t ws_size,
                              hipStream_t stream) {
  const int* node_idx  = (const int*)d_in[0];
  const int* group_idx = (const int*)d_in[1];
  const float* node_embedding  = (const float*)d_in[2];
  const float* node_vec        = (const float*)d_in[3];
  const float* group_embedding = (const float*)d_in[4];
  const float* group_vec       = (const float*)d_in[5];
  const float* edge_attr       = (const float*)d_in[6];
  // d_in[7] = edge_weight (unused by reference)
  const float* edge_vec        = (const float*)d_in[8];
  const float* Wq = (const float*)d_in[9],  *bq = (const float*)d_in[10];
  const float* Wk = (const float*)d_in[11], *bk = (const float*)d_in[12];
  const float* Wv = (const float*)d_in[13], *bv = (const float*)d_in[14];
  const float* Wp1 = (const float*)d_in[15], *bp1 = (const float*)d_in[16];
  const float* Wp2 = (const float*)d_in[17], *bp2 = (const float*)d_in[18];
  const float* Ws1 = (const float*)d_in[19], *bs1 = (const float*)d_in[20];
  const float* Ws2 = (const float*)d_in[21], *bs2 = (const float*)d_in[22];
  const float* L0 = (const float*)d_in[23];
  const float* L1 = (const float*)d_in[24];
  const float* L2 = (const float*)d_in[25];
  const float* L3 = (const float*)d_in[26];
  const float* L4 = (const float*)d_in[27];
  const float* L5 = (const float*)d_in[28];

  const int H = H_DIM;
  const int N = in_sizes[2] / H;
  const int G = in_sizes[4] / H;
  const int E = in_sizes[0];
  const size_t NH = (size_t)N * H;

  // ---- workspace: fp32 section first, then 16B-aligned bf16 section ------
  float* fws = (float*)d_ws;
  size_t fo = 0;
  float* msnode = fws + fo; fo += NH;
  float* t0     = fws + fo; fo += NH;
  float* t4     = fws + fo; fo += NH;
  float* t5     = fws + fo; fo += NH;
  float* sbuf   = fws + fo; fo += NH;

  __bf16* bws = (__bf16*)(fws + fo);
  size_t bo = 0;
  __bf16* qbuf  = bws + bo; bo += NH;
  __bf16* kbuf  = bws + bo; bo += (size_t)G * H;
  __bf16* vbuf  = bws + bo; bo += (size_t)G * H;
  __bf16* msbuf = bws + bo; bo += (size_t)E * H;
  __bf16* h1buf = bws + bo; bo += (size_t)E * (H / 2);
  __bf16* spbuf = bws + bo; bo += (size_t)E * H;
  __bf16* svbuf = bws + bo; bo += (size_t)E * H;
  __bf16* nvb   = bws + bo; bo += (size_t)3 * NH;  // node_vec staged bf16
  __bf16* msnb  = bws + bo; bo += NH;              // m_s_node staged bf16
  // packed weights (bf16, B-frag layout)
  __bf16* pWq  = bws + bo; bo += (size_t)H * H;
  __bf16* pWk  = bws + bo; bo += (size_t)H * H;
  __bf16* pWv  = bws + bo; bo += (size_t)H * H;
  __bf16* pWp1 = bws + bo; bo += (size_t)H * (H / 2);
  __bf16* pWp2 = bws + bo; bo += (size_t)(H / 2) * H;
  __bf16* pWs1 = bws + bo; bo += (size_t)H * (H / 2);
  __bf16* pWs2 = bws + bo; bo += (size_t)(H / 2) * H;
  __bf16* pL0  = bws + bo; bo += (size_t)H * H;
  __bf16* pL1  = bws + bo; bo += (size_t)H * H;
  __bf16* pL2  = bws + bo; bo += (size_t)H * H;
  __bf16* pL3  = bws + bo; bo += (size_t)H * H;
  __bf16* pL4  = bws + bo; bo += (size_t)H * H;
  __bf16* pL5  = bws + bo; bo += (size_t)H * H;

  float* dx = (float*)d_out;
  float* dv = dx + NH;

  const int BT = 256;                       // 8 waves / block
  auto blk = [](long waves) { return (int)((waves + 7) / 8); };

  // 0) zero accumulators
  {
    size_t nz = (size_t)N * 3 * H;
    zero_kernel<<<(int)((nz + BT - 1) / BT), BT, 0, stream>>>(dv, nz);
    zero_kernel<<<(int)((NH + BT - 1) / BT), BT, 0, stream>>>(msnode, NH);
  }

  // 1) pack weights; stage node_vec -> bf16
  {
    const long wHH = (long)(H / 32) * (H / 16);
    const long wH64 = (long)(H / 32) * ((H / 2) / 16);
    const long w64H = (long)((H / 2) / 32) * (H / 16);
    pack_w_kernel<<<blk(wHH), BT, 0, stream>>>(Wq, pWq, H, H);
    pack_w_kernel<<<blk(wHH), BT, 0, stream>>>(Wk, pWk, H, H);
    pack_w_kernel<<<blk(wHH), BT, 0, stream>>>(Wv, pWv, H, H);
    pack_w_kernel<<<blk(wH64), BT, 0, stream>>>(Wp1, pWp1, H, H / 2);
    pack_w_kernel<<<blk(w64H), BT, 0, stream>>>(Wp2, pWp2, H / 2, H);
    pack_w_kernel<<<blk(wH64), BT, 0, stream>>>(Ws1, pWs1, H, H / 2);
    pack_w_kernel<<<blk(w64H), BT, 0, stream>>>(Ws2, pWs2, H / 2, H);
    pack_w_kernel<<<blk(wHH), BT, 0, stream>>>(L0, pL0, H, H);
    pack_w_kernel<<<blk(wHH), BT, 0, stream>>>(L1, pL1, H, H);
    pack_w_kernel<<<blk(wHH), BT, 0, stream>>>(L2, pL2, H, H);
    pack_w_kernel<<<blk(wHH), BT, 0, stream>>>(L3, pL3, H, H);
    pack_w_kernel<<<blk(wHH), BT, 0, stream>>>(L4, pL4, H, H);
    pack_w_kernel<<<blk(wHH), BT, 0, stream>>>(L5, pL5, H, H);

    const size_t n8 = (size_t)3 * NH / 8;
    cvt_f32_bf16_kernel<<<(int)((n8 + BT - 1) / BT), BT, 0, stream>>>(
        node_vec, nvb, n8);
  }

  // 2) q/k/val projections: fp32 A -> bf16 C, +bias
  const long tQ = ((long)N / 32) * (H / 64);        // ROW2 (N % 32 == 0)
  const long tG = ((long)G / 16) * (H / 64);        // 16-row (G % 32 != 0)
  wmma_gemm4_kernel<true, false, true, true, false><<<blk(tQ), BT, 0, stream>>>(
      node_embedding, pWq, bq, qbuf, N, H, H);
  wmma_gemm4_kernel<false, false, true, true, false><<<blk(tG), BT, 0, stream>>>(
      group_embedding, pWk, bk, kbuf, G, H, H);
  wmma_gemm4_kernel<false, false, true, true, false><<<blk(tG), BT, 0, stream>>>(
      group_embedding, pWv, bv, vbuf, G, H, H);

  // 3) edge attention -> m_s (bf16) + m_s_node scatter (f32 atomics)
  edge_attn_kernel<<<(E + 7) / 8, BT, 0, stream>>>(
      node_idx, group_idx, qbuf, kbuf, vbuf, edge_attr, msbuf, msnode, E);

  // 4) edge MLPs (H -> H/2 SiLU -> H), all-bf16 activation path, ROW2
  const long tE1 = ((long)E / 32) * ((H / 2) / 64);
  const long tE2 = ((long)E / 32) * (H / 64);
  wmma_gemm4_kernel<true, true, true, true, true><<<blk(tE1), BT, 0, stream>>>(
      msbuf, pWp1, bp1, h1buf, E, H, H / 2);
  wmma_gemm4_kernel<true, true, true, true, false><<<blk(tE2), BT, 0, stream>>>(
      h1buf, pWp2, bp2, spbuf, E, H / 2, H);
  wmma_gemm4_kernel<true, true, true, true, true><<<blk(tE1), BT, 0, stream>>>(
      msbuf, pWs1, bs1, h1buf, E, H, H / 2);
  wmma_gemm4_kernel<true, true, true, true, false><<<blk(tE2), BT, 0, stream>>>(
      h1buf, pWs2, bs2, svbuf, E, H / 2, H);

  // 5) m_v scatter-add into dv part of d_out
  edge_mv_kernel<<<(E + 7) / 8, BT, 0, stream>>>(
      node_idx, group_idx, spbuf, svbuf, edge_vec, group_vec, dv, E);

  // 6) stage m_s_node -> bf16, then fused tri-GEMM t0/t4/t5
  {
    const size_t n8 = NH / 8;
    cvt_f32_bf16_kernel<<<(int)((n8 + BT - 1) / BT), BT, 0, stream>>>(
        msnode, msnb, n8);
    const long tT = ((long)N / 16) * (H / 64);
    wmma_gemm_tri_kernel<<<blk(tT), BT, 0, stream>>>(
        msnb, pL0, pL4, pL5, t0, t4, t5, N);
  }

  // 7) fused (node_vec@L2 * node_vec@L3).sum(axis=1)
  {
    const long tV = ((long)N / 16) * (H / 64);
    v1v2_wmma_kernel<<<blk(tV), BT, 0, stream>>>(nvb, pL2, pL3, sbuf, N);
  }

  // 8) dx = s * t4 + t5
  final_dx_kernel<<<(int)((NH + BT - 1) / BT), BT, 0, stream>>>(
      sbuf, t4, t5, dx, NH);

  // 9) dv += (m_s_node@L0)[:,None,:] * (node_vec@L1)
  {
    const long tDV = ((long)(3 * N) / 16) * (H / 64);
    wmma_gemm_rowscale_add_kernel<<<blk(tDV), BT, 0, stream>>>(
        nvb, pL1, t0, dv, 3 * N, H, H);
  }
}
